// SetAbstractionMSG_3358664425517
// MI455X (gfx1250) — matrix operand order repaired and checked by reading, not compiled, not run on hardware
//
#include <hip/hip_runtime.h>
#include <cstdint>
#include <cstddef>

#define NPTS   4096
#define NPOINT 1024
#define EPS_BN 1e-5f

typedef __attribute__((ext_vector_type(8)))  unsigned short v8us;
typedef __attribute__((ext_vector_type(16))) unsigned short v16us;
typedef __attribute__((ext_vector_type(16))) __bf16        v16bf;
typedef __attribute__((ext_vector_type(8)))  float         v8f;

__device__ __forceinline__ unsigned short f2bf(float f) {
    unsigned u = __float_as_uint(f);
    unsigned r = u + 0x7FFFu + ((u >> 16) & 1u);   // round-to-nearest-even
    return (unsigned short)(r >> 16);
}
__device__ __forceinline__ float bf2f(unsigned short h) {
    return __uint_as_float(((unsigned)h) << 16);
}

// ---------------- FPS: one workgroup per batch, cloud resident in LDS --------
__global__ __launch_bounds__(1024) void fps_kernel(const float* __restrict__ xyz,
                                                   float* __restrict__ newxyz) {
    __shared__ float px[NPTS], py[NPTS], pz[NPTS], dist[NPTS];
    __shared__ float redv[32];
    __shared__ int   redi[32];
    __shared__ int   sFar;
    const int b = blockIdx.x, t = threadIdx.x;
    const int lane = t & 31, wave = t >> 5;
    const float* base = xyz + (size_t)b * NPTS * 3;
    for (int i = t; i < NPTS; i += 1024) {
        px[i] = base[i * 3 + 0];
        py[i] = base[i * 3 + 1];
        pz[i] = base[i * 3 + 2];
        dist[i] = 1e10f;
    }
    __syncthreads();
    int far = 0;
    for (int iter = 0; iter < NPOINT; ++iter) {
        const float cx = px[far], cy = py[far], cz = pz[far];
        float bv = -1.0f; int bi = 0;
#pragma unroll
        for (int j = 0; j < 4; ++j) {
            int i = t + j * 1024;
            float dx = px[i] - cx, dy = py[i] - cy, dz = pz[i] - cz;
            float nd = fminf(dist[i], dx * dx + dy * dy + dz * dz);
            dist[i] = nd;
            if (nd > bv) { bv = nd; bi = i; }
        }
#pragma unroll
        for (int off = 16; off > 0; off >>= 1) {
            float ov = __shfl_down(bv, off, 32);
            int   oi = __shfl_down(bi, off, 32);
            if (ov > bv) { bv = ov; bi = oi; }
        }
        if (lane == 0) { redv[wave] = bv; redi[wave] = bi; }
        __syncthreads();
        if (t < 32) {
            bv = redv[t]; bi = redi[t];
#pragma unroll
            for (int off = 16; off > 0; off >>= 1) {
                float ov = __shfl_down(bv, off, 32);
                int   oi = __shfl_down(bi, off, 32);
                if (ov > bv) { bv = ov; bi = oi; }
            }
            if (t == 0) {
                sFar = bi;
                float* o = newxyz + ((size_t)b * NPOINT + iter) * 3;
                o[0] = cx; o[1] = cy; o[2] = cz;   // centroid at start of step
            }
        }
        __syncthreads();
        far = sFar;
    }
}

// ------------- ball query: one wave per query, streaming index order ---------
__global__ __launch_bounds__(256) void ballq_kernel(const float* __restrict__ xyz,
                                                    const float* __restrict__ newxyz,
                                                    int* __restrict__ gidx,
                                                    float r2, int K) {
    const int w = blockIdx.x * 8 + (threadIdx.x >> 5);   // b*1024+s
    const int lane = threadIdx.x & 31;
    const int b = w >> 10;
    const float* q = newxyz + (size_t)w * 3;
    const float qx = q[0], qy = q[1], qz = q[2];
    const float* base = xyz + (size_t)b * NPTS * 3;
    int cnt = 0, first = 0;
    for (int n0 = 0; n0 < NPTS && cnt < K; n0 += 32) {
        int n = n0 + lane;
        float dx = base[n * 3 + 0] - qx;
        float dy = base[n * 3 + 1] - qy;
        float dz = base[n * 3 + 2] - qz;
        bool in = (dx * dx + dy * dy + dz * dz) <= r2;
        unsigned m = (unsigned)__ballot(in);
        while (m && cnt < K) {                 // m is wave-uniform
            int j = __builtin_ctz(m); m &= m - 1;
            int idx = n0 + j;
            if (cnt == 0) first = idx;
            if (lane == 0) gidx[(size_t)w * K + cnt] = idx;
            cnt++;
        }
    }
    for (int i = cnt + lane; i < K; i += 32) gidx[(size_t)w * K + i] = first;
}

// -------- gather + concat(xyz-rel, feats) + pad 67->96, store bf16 -----------
__global__ __launch_bounds__(256) void gather_kernel(const float* __restrict__ xyz,
                                                     const float* __restrict__ pts,
                                                     const float* __restrict__ newxyz,
                                                     const int* __restrict__ gidx,
                                                     unsigned short* __restrict__ X,
                                                     int K, long total) {
    long t = (long)blockIdx.x * 256 + threadIdx.x;
    if (t >= total) return;
    int  c = (int)(t % 96);
    long r = t / 96;
    int gi = gidx[r];
    int bs = (int)(r / K);          // b*1024+s
    int b = bs >> 10;
    float v;
    if (c < 3)       v = xyz[((size_t)b * NPTS + gi) * 3 + c] - newxyz[(size_t)bs * 3 + c];
    else if (c < 67) v = pts[((size_t)b * NPTS + gi) * 64 + (c - 3)];
    else             v = 0.0f;
    X[(size_t)r * 96 + c] = f2bf(v);
}

// ---------------- pack W [Cout,Cin] f32 -> [Cout,Cpad] bf16 ------------------
__global__ __launch_bounds__(256) void packw_kernel(const float* __restrict__ W,
                                                    unsigned short* __restrict__ Wp,
                                                    int Cin, int Cpad, int total) {
    int t = blockIdx.x * 256 + threadIdx.x;
    if (t >= total) return;
    int c = t % Cpad, o = t / Cpad;
    Wp[t] = (c < Cin) ? f2bf(W[o * Cin + c]) : (unsigned short)0;
}

// ------------- WMMA GEMM: Y[R,Cout] = X[R,Cpad] * Wp^T + bias ----------------
// Each wave owns one 16-row stripe and ALL NTC = Cout/16 column tiles:
// the A fragment is loaded once per k-step and reused by NTC WMMAs, so the
// activation matrix is read from memory exactly once per layer. Weights
// (<= 24 KB) stay hot in L1. Bias is folded into the accumulator init.
template <int NTC>
__global__ __launch_bounds__(128) void gemm_kernel(const unsigned short* __restrict__ X,
                                                   const unsigned short* __restrict__ Wp,
                                                   const float* __restrict__ bias,
                                                   unsigned short* __restrict__ Y,
                                                   int Cpad, int Cout, int ntR) {
    const int wid = blockIdx.x * 4 + (threadIdx.x >> 5);
    if (wid >= ntR) return;                   // wave-uniform: EXEC all-1s for WMMA
    const int lane = threadIdx.x & 31;
    const int m = lane & 15, half = lane >> 4;
    const unsigned short* xrow = X + (size_t)(wid * 16 + m) * Cpad;
    v8f acc[NTC];
#pragma unroll
    for (int ct = 0; ct < NTC; ++ct) {        // D element v = (row half*8+v, col ct*16+m)
        float bv = bias[ct * 16 + m];
        acc[ct] = (v8f){bv, bv, bv, bv, bv, bv, bv, bv};
    }
    for (int k0 = 0; k0 < Cpad; k0 += 32) {
        // A 16x32 bf16: lane(m,half) holds K = half*16..+15 of row m
        v16us au = *(const v16us*)(xrow + k0 + half * 16);
        v16bf A = __builtin_bit_cast(v16bf, au);
#pragma unroll
        for (int ct = 0; ct < NTC; ++ct) {
            // B 32x16 bf16: lane n holds col n, K = half*16..+15 (W^T tile = W row)
            v16us bu = *(const v16us*)(Wp + (size_t)(ct * 16 + m) * Cpad + k0 + half * 16);
            acc[ct] = __builtin_amdgcn_wmma_f32_16x16x32_bf16(
                false, A, false, __builtin_bit_cast(v16bf, bu),
                (short)0, acc[ct], false, false);
        }
    }
    unsigned short* yo = Y + (size_t)(wid * 16 + half * 8) * Cout + m;
#pragma unroll
    for (int ct = 0; ct < NTC; ++ct)
#pragma unroll
        for (int v = 0; v < 8; ++v)
            yo[(size_t)v * Cout + ct * 16] = f2bf(acc[ct][v]);
}

// ---------------- per-channel batch stats (sum, sumsq) -----------------------
__global__ void zstat_kernel(float* st) { st[threadIdx.x] = 0.0f; }

__global__ void stats_kernel(const unsigned short* __restrict__ Y,
                             float* __restrict__ st, int R, int Cout) {
    const int c = threadIdx.x;                // blockDim.x == Cout (coalesced reads)
    float s1 = 0.f, s2 = 0.f;
    for (int r = blockIdx.x; r < R; r += gridDim.x) {
        float y = bf2f(Y[(size_t)r * Cout + c]);
        s1 += y; s2 += y * y;
    }
    atomicAdd(&st[c], s1);
    atomicAdd(&st[128 + c], s2);
}

// ---------------- BN (batch stats) + ReLU, write next-layer input ------------
__global__ __launch_bounds__(256) void bnrelu_kernel(const unsigned short* __restrict__ Y,
                                                     unsigned short* __restrict__ Xn,
                                                     const float* __restrict__ st,
                                                     const float* __restrict__ gamma,
                                                     const float* __restrict__ beta,
                                                     long total, int Cout, float invR) {
    long t = (long)blockIdx.x * 256 + threadIdx.x;
    if (t >= total) return;
    int c = (int)(t % Cout);
    float mean = st[c] * invR;
    float var  = st[128 + c] * invR - mean * mean;
    float y = bf2f(Y[t]);
    float o = gamma[c] * (y - mean) * rsqrtf(var + EPS_BN) + beta[c];
    Xn[t] = f2bf(fmaxf(o, 0.0f));
}

// ---------------- max over K into output channel slice -----------------------
__global__ __launch_bounds__(256) void maxpool_kernel(const unsigned short* __restrict__ Xn,
                                                      float* __restrict__ out,
                                                      int K, int C, int choff, int total) {
    int t = blockIdx.x * 256 + threadIdx.x;
    if (t >= total) return;
    int c = t % C, g = t / C;
    float m = -3.4e38f;
    for (int k = 0; k < K; ++k)
        m = fmaxf(m, bf2f(Xn[((size_t)g * K + k) * C + c]));
    out[(size_t)g * 320 + choff + c] = m;
}

extern "C" void kernel_launch(void* const* d_in, const int* in_sizes, int n_in,
                              void* d_out, int out_size, void* d_ws, size_t ws_size,
                              hipStream_t stream) {
    (void)in_sizes; (void)n_in; (void)out_size; (void)ws_size;
    const float* xyz = (const float*)d_in[0];
    const float* pts = (const float*)d_in[1];
    float* out    = (float*)d_out;
    float* newxyz = out;                 // 8*1024*3
    float* newpts = out + 8 * 1024 * 3;  // 8*1024*320

    // workspace layout (bytes, 256-aligned)
    char* ws = (char*)d_ws;
    int*            gidx = (int*)ws;                                    //   2,097,152
    unsigned short* Xb   = (unsigned short*)(ws + 2097152u);            // 100,663,296
    unsigned short* Yb   = (unsigned short*)(ws + 102760448u);          // 134,217,728
    unsigned short* Wpb  = (unsigned short*)(ws + 236978176u);          //      32,768
    float*          st   = (float*)(ws + 237010944u);                   //       1,024

    fps_kernel<<<8, 1024, 0, stream>>>(xyz, newxyz);

    static const int   KS[3]      = {16, 32, 64};
    static const float R2v[3]     = {0.01f, 0.04f, 0.16f};
    static const int   CIN[3][3]  = {{67, 32, 32}, {67, 64, 64}, {67, 64, 96}};
    static const int   CPAD[3][3] = {{96, 32, 32}, {96, 64, 64}, {96, 64, 96}};
    static const int   COUT[3][3] = {{32, 32, 64}, {64, 64, 128}, {64, 96, 128}};
    static const int   CHOFF[3]   = {0, 64, 192};

    for (int s = 0; s < 3; ++s) {
        const int K = KS[s];
        const long R = 8L * 1024 * K;
        ballq_kernel<<<1024, 256, 0, stream>>>(xyz, newxyz, gidx, R2v[s], K);
        long tg = R * 96;
        gather_kernel<<<(int)((tg + 255) / 256), 256, 0, stream>>>(xyz, pts, newxyz,
                                                                   gidx, Xb, K, tg);
        for (int l = 0; l < 3; ++l) {
            const int Cin = CIN[s][l], Cpad = CPAD[s][l], Cout = COUT[s][l];
            const int pbase = 2 + (s * 3 + l) * 4;
            const float* W     = (const float*)d_in[pbase + 0];
            const float* bias  = (const float*)d_in[pbase + 1];
            const float* gamma = (const float*)d_in[pbase + 2];
            const float* beta  = (const float*)d_in[pbase + 3];
            int wtotal = Cout * Cpad;
            packw_kernel<<<(wtotal + 255) / 256, 256, 0, stream>>>(W, Wpb, Cin, Cpad, wtotal);
            const int ntR = (int)(R / 16);
            const int ntC = Cout / 16;
            const int blocks = (ntR + 3) / 4;
            switch (ntC) {
            case 2: gemm_kernel<2><<<blocks, 128, 0, stream>>>(Xb, Wpb, bias, Yb, Cpad, Cout, ntR); break;
            case 4: gemm_kernel<4><<<blocks, 128, 0, stream>>>(Xb, Wpb, bias, Yb, Cpad, Cout, ntR); break;
            case 6: gemm_kernel<6><<<blocks, 128, 0, stream>>>(Xb, Wpb, bias, Yb, Cpad, Cout, ntR); break;
            default: gemm_kernel<8><<<blocks, 128, 0, stream>>>(Xb, Wpb, bias, Yb, Cpad, Cout, ntR); break;
            }
            zstat_kernel<<<1, 256, 0, stream>>>(st);
            stats_kernel<<<256, Cout, 0, stream>>>(Yb, st, (int)R, Cout);
            long te = R * Cout;
            bnrelu_kernel<<<(int)((te + 255) / 256), 256, 0, stream>>>(Yb, Xb, st, gamma, beta,
                                                                       te, Cout, 1.0f / (float)R);
        }
        int mp = 8192 * COUT[s][2];
        maxpool_kernel<<<(mp + 255) / 256, 256, 0, stream>>>(Xb, newpts, K, COUT[s][2],
                                                             CHOFF[s], mp);
    }
}
